// SelfAttention_592705487560
// MI455X (gfx1250) — compile-verified, matrix operand order checked by reading, not run on hardware
//
#include <hip/hip_runtime.h>
#include <hip/hip_bf16.h>

typedef _Float16 f16;
typedef __attribute__((ext_vector_type(16))) _Float16 v16h;
typedef __attribute__((ext_vector_type(8)))  _Float16 v8h;
typedef __attribute__((ext_vector_type(8)))  float    v8f;

#define N_SEQ 4096
#define DIM   128
#define NH    8

__device__ __forceinline__ v16h pack2(v8h lo, v8h hi) {
  v16h r;
#pragma unroll
  for (int i = 0; i < 8; i++) { r[i] = lo[i]; r[i + 8] = hi[i]; }
  return r;
}

// One 16-f16-per-lane fragment: 8 contiguous f16 at base, 8 more at base+16.
__device__ __forceinline__ v16h load_frag_f16(const f16* base) {
  return pack2(*(const v8h*)base, *(const v8h*)(base + 16));
}

__device__ __forceinline__ v8f wmma16(v16h a, v16h b, v8f c) {
  return __builtin_amdgcn_wmma_f32_16x16x32_f16(false, a, false, b, (short)0, c,
                                                false, false);
}

// Reductions across the 16-lane half holding one matrix row (wave32).
__device__ __forceinline__ float redmax16(float v) {
#pragma unroll
  for (int m = 1; m < 16; m <<= 1) v = fmaxf(v, __shfl_xor(v, m, 32));
  return v;
}
__device__ __forceinline__ float redsum16(float v) {
#pragma unroll
  for (int m = 1; m < 16; m <<= 1) v += __shfl_xor(v, m, 32);
  return v;
}

// ---------------------------------------------------------------------------
// Kernel 0: data prep. x -> f16; Wq/Wk/Wv -> f16 transposed [p,h][e][d];
// Wo -> f16 transposed [e][k]. Makes every later fragment load a b128.
// Exactly 1048576 threads: 524288 (x) + 393216 (Wqkv) + 131072 (Wo).
// ---------------------------------------------------------------------------
__global__ __launch_bounds__(256) void prep_kernel(
    const float* __restrict__ x,
    const float* __restrict__ Wq, const float* __restrict__ Wk,
    const float* __restrict__ Wv, const float* __restrict__ Wo,
    f16* __restrict__ x16, f16* __restrict__ WT, f16* __restrict__ WoT) {
  int tid = blockIdx.x * 256 + threadIdx.x;
  if (tid < N_SEQ * DIM) {               // x copy-convert (coalesced)
    x16[tid] = (f16)x[tid];
    return;
  }
  tid -= N_SEQ * DIM;
  if (tid < 3 * NH * DIM * DIM) {        // W transpose: read coalesced over e
    const int e = tid & 127; int t = tid >> 7;
    const int d = t & 127;   t >>= 7;
    const int h = t & 7;
    const int p = t >> 3;
    const float* W = (p == 0) ? Wq : (p == 1) ? Wk : Wv;
    WT[((size_t)(p * NH + h) * DIM + e) * DIM + d] =
        (f16)W[((size_t)h * DIM + d) * DIM + e];
    return;
  }
  tid -= 3 * NH * DIM * DIM;             // Wo transpose [1024][128] -> [128][1024]
  const int e = tid & 127;
  const int k = tid >> 7;
  WoT[(size_t)e * (NH * DIM) + k] = (f16)Wo[(size_t)k * DIM + e];
}

// ---------------------------------------------------------------------------
// Kernel 1a: q and v projections (row-major outputs [H][N][D]).
// One wave: 16 rows x all 128 cols for one (proj, head, n-tile).
// Destination chosen once -> unconditional stores, constant offsets.
// Wave-tiles: 2 * 8 * 256 = 4096 -> 512 blocks.
// ---------------------------------------------------------------------------
__global__ __launch_bounds__(256) void qv_proj_kernel(
    const f16* __restrict__ x16, const f16* __restrict__ WT,
    const float* __restrict__ bq, const float* __restrict__ bv,
    f16* __restrict__ q16, f16* __restrict__ v16) {
  const int lane = threadIdx.x & 31;
  const int wave = threadIdx.x >> 5;
  const int llo = lane & 15, lhi = lane >> 4;
  int tile = blockIdx.x * 8 + wave;
  const int ntile = tile & 255; tile >>= 8;
  const int h = tile & 7;
  const int pv = tile >> 3;  // 0=q 1=v

  const float* B = pv ? bv : bq;
  const f16* Wt = WT + (size_t)((pv ? 2 : 0) * NH + h) * DIM * DIM;  // [e][d]
  f16* const dsth = (pv ? v16 : q16) + (size_t)h * N_SEQ * DIM;
  const int n0 = ntile * 16;

  v16h a[4];
  {
    const f16* ab = x16 + (size_t)(n0 + llo) * DIM + lhi * 8;
#pragma unroll
    for (int kc = 0; kc < 4; kc++) a[kc] = load_frag_f16(ab + kc * 32);
  }

#pragma unroll
  for (int et = 0; et < 8; et++) {
    const int e0 = et * 16;
    v8f acc = {};
    const f16* wb = Wt + (size_t)(e0 + llo) * DIM + lhi * 8;
#pragma unroll
    for (int kc = 0; kc < 4; kc++)
      acc = wmma16(a[kc], load_frag_f16(wb + kc * 32), acc);
    const float bias = B[h * DIM + e0 + llo];
    f16* p = dsth + (size_t)(n0 + lhi * 8) * DIM + e0 + llo;
#pragma unroll
    for (int j = 0; j < 8; j++)
      p[(size_t)j * DIM] = (f16)(acc[j] + bias);
  }
}

// ---------------------------------------------------------------------------
// Kernel 1b: k projection, stored transposed kT16 [H][D][N].
// A lane's 8 accumulator rows are contiguous in the transposed layout ->
// one packed global_store_b128 per e-tile.
// Wave-tiles: 8 * 256 = 2048 -> 256 blocks.
// ---------------------------------------------------------------------------
__global__ __launch_bounds__(256) void k_proj_kernel(
    const f16* __restrict__ x16, const f16* __restrict__ WT,
    const float* __restrict__ bk, f16* __restrict__ kT16) {
  const int lane = threadIdx.x & 31;
  const int wave = threadIdx.x >> 5;
  const int llo = lane & 15, lhi = lane >> 4;
  const int tile = blockIdx.x * 8 + wave;
  const int ntile = tile & 255;
  const int h = tile >> 8;

  const f16* Wt = WT + (size_t)(1 * NH + h) * DIM * DIM;  // [e][d]
  const int n0 = ntile * 16;

  v16h a[4];
  {
    const f16* ab = x16 + (size_t)(n0 + llo) * DIM + lhi * 8;
#pragma unroll
    for (int kc = 0; kc < 4; kc++) a[kc] = load_frag_f16(ab + kc * 32);
  }

#pragma unroll
  for (int et = 0; et < 8; et++) {
    const int e0 = et * 16;
    v8f acc = {};
    const f16* wb = Wt + (size_t)(e0 + llo) * DIM + lhi * 8;
#pragma unroll
    for (int kc = 0; kc < 4; kc++)
      acc = wmma16(a[kc], load_frag_f16(wb + kc * 32), acc);
    const float bias = bk[h * DIM + e0 + llo];
    v8h pk;
#pragma unroll
    for (int j = 0; j < 8; j++) pk[j] = (f16)(acc[j] + bias);
    *(v8h*)(kT16 + (size_t)(h * DIM + e0 + llo) * N_SEQ + n0 + lhi * 8) = pk;
  }
}

// ---------------------------------------------------------------------------
// Kernel 2: streaming-softmax attention.
// scores[n,m] = sum_d v[n,d] q[m,d]; P = softmax_m; out[n,d] = sum_m P[n,m] k[m,d].
// One wave owns a 32-row tile of n (two 16-row sub-tiles sharing all q/kT
// B-fragments -> 16 B/lane/WMMA), streams m in blocks of 32.
// Wave-tiles: 8 heads * 128 = 1024 -> 128 blocks.
// ---------------------------------------------------------------------------
__global__ __launch_bounds__(256) void flash_attn_kernel(
    const f16* __restrict__ q16, const f16* __restrict__ kT16,
    const f16* __restrict__ v16, f16* __restrict__ outs) {
  const int lane = threadIdx.x & 31;
  const int wave = threadIdx.x >> 5;
  const int llo = lane & 15, lhi = lane >> 4;
  const int gt = blockIdx.x * 8 + wave;   // 0..1023
  const int h  = gt >> 7;
  const int m0 = (gt & 127) * 32;         // row-tile base (n index)

  __shared__ __align__(16) f16 plds[8][2][16 * 32];  // per-wave P scratch (16KB)

  const f16* __restrict__ vh = v16  + (size_t)h * N_SEQ * DIM;
  const f16* __restrict__ qh = q16  + (size_t)h * N_SEQ * DIM;
  const f16* __restrict__ kh = kT16 + (size_t)h * DIM * N_SEQ;

  // A fragments: 2 x 16 rows of v, full K=128
  v16h av[2][4];
#pragma unroll
  for (int r = 0; r < 2; r++) {
    const f16* base = vh + (size_t)(m0 + r * 16 + llo) * DIM + lhi * 8;
#pragma unroll
    for (int kc = 0; kc < 4; kc++) av[r][kc] = load_frag_f16(base + kc * 32);
  }

  const v8f vzero = {};
  float mrun[2][8], lrun[2][8];
  v8f O[2][8];
#pragma unroll
  for (int r = 0; r < 2; r++) {
#pragma unroll
    for (int j = 0; j < 8; j++) { mrun[r][j] = -3.0e38f; lrun[r][j] = 0.f; }
#pragma unroll
    for (int dt = 0; dt < 8; dt++) O[r][dt] = vzero;
  }

  for (int nb = 0; nb < N_SEQ; nb += 32) {
    // ---- load q B-fragments once, shared by both row tiles ----
    v16h bq[2][4];
#pragma unroll
    for (int t = 0; t < 2; t++) {
      const f16* qb = qh + (size_t)(nb + t * 16 + llo) * DIM + lhi * 8;
#pragma unroll
      for (int kc = 0; kc < 4; kc++) bq[t][kc] = load_frag_f16(qb + kc * 32);
    }

    // ---- S = v-tiles @ q-tiles^T : 2x2 16x16 tiles, K=128 ----
    v8f S[2][2];
#pragma unroll
    for (int r = 0; r < 2; r++)
#pragma unroll
      for (int t = 0; t < 2; t++) {
        v8f s = {};
#pragma unroll
        for (int kc = 0; kc < 4; kc++) s = wmma16(av[r][kc], bq[t][kc], s);
        S[r][t] = s;
      }

    // ---- online softmax over the 32 new columns, per row tile ----
    float alpha[2][8];
#pragma unroll
    for (int r = 0; r < 2; r++)
#pragma unroll
      for (int j = 0; j < 8; j++) {
        const float rx   = redmax16(fmaxf(S[r][0][j], S[r][1][j]));
        const float mnew = fmaxf(mrun[r][j], rx);
        const float a    = __expf(mrun[r][j] - mnew);
        const float p0   = __expf(S[r][0][j] - mnew);
        const float p1   = __expf(S[r][1][j] - mnew);
        S[r][0][j] = p0; S[r][1][j] = p1;
        lrun[r][j] = lrun[r][j] * a + redsum16(p0 + p1);
        mrun[r][j] = mnew;
        alpha[r][j] = a;
      }

    // ---- rotate P (C-layout) -> A-fragments via per-wave LDS ----
#pragma unroll
    for (int r = 0; r < 2; r++)
#pragma unroll
      for (int t = 0; t < 2; t++)
#pragma unroll
        for (int j = 0; j < 8; j++)
          plds[wave][r][(j + lhi * 8) * 32 + t * 16 + llo] = (f16)S[r][t][j];
    asm volatile("" ::: "memory");

    v16h pf[2];
#pragma unroll
    for (int r = 0; r < 2; r++)
      pf[r] = load_frag_f16(plds[wave][r] + llo * 32 + lhi * 8);

    // ---- O = O*alpha + P @ k-block; kT B-fragment shared by both row tiles ----
#pragma unroll
    for (int dt = 0; dt < 8; dt++) {
      const v16h b =
          load_frag_f16(kh + (size_t)(dt * 16 + llo) * N_SEQ + nb + lhi * 8);
#pragma unroll
      for (int r = 0; r < 2; r++) {
        v8f o = O[r][dt];
#pragma unroll
        for (int j = 0; j < 8; j++) o[j] *= alpha[r][j];
        O[r][dt] = wmma16(pf[r], b, o);
      }
    }
    asm volatile("" ::: "memory");  // keep next iter's LDS writes after reads
  }

  // ---- normalize and store into concat layout [N][H*D], head-major cols ----
#pragma unroll
  for (int r = 0; r < 2; r++) {
    float inv[8];
#pragma unroll
    for (int j = 0; j < 8; j++) inv[j] = __builtin_amdgcn_rcpf(lrun[r][j]);
#pragma unroll
    for (int dt = 0; dt < 8; dt++)
#pragma unroll
      for (int j = 0; j < 8; j++) {
        const int n = m0 + r * 16 + j + lhi * 8;
        outs[(size_t)n * (NH * DIM) + h * DIM + dt * 16 + llo] =
            (f16)(O[r][dt][j] * inv[j]);
      }
  }
}

// ---------------------------------------------------------------------------
// Kernel 3: out = outs[N,1024] @ Wo[1024,128] + bo, f32 result.
// B fragments from pre-transposed WoT[e][k] -> all loads are b128.
// ---------------------------------------------------------------------------
__global__ __launch_bounds__(256) void out_proj_kernel(
    const f16* __restrict__ outs, const f16* __restrict__ WoT,
    const float* __restrict__ bo, float* __restrict__ out) {
  const int lane = threadIdx.x & 31;
  const int wave = threadIdx.x >> 5;
  const int llo = lane & 15, lhi = lane >> 4;
  const int tile = blockIdx.x * 8 + wave;
  const int etile = tile & 7;
  const int ntile = tile >> 3;
  const int n0 = ntile * 16, e0 = etile * 16;

  const f16* ab = outs + (size_t)(n0 + llo) * (NH * DIM) + lhi * 8;
  const f16* wb = WoT + (size_t)(e0 + llo) * (NH * DIM) + lhi * 8;

  v8f acc = {};
#pragma unroll 4
  for (int kc = 0; kc < 32; kc++) {
    acc = wmma16(load_frag_f16(ab + kc * 32), load_frag_f16(wb + kc * 32), acc);
  }
  const float bias = bo[e0 + llo];
#pragma unroll
  for (int j = 0; j < 8; j++)
    out[(size_t)(n0 + j + lhi * 8) * DIM + e0 + llo] = acc[j] + bias;
}

// ---------------------------------------------------------------------------
extern "C" void kernel_launch(void* const* d_in, const int* in_sizes, int n_in,
                              void* d_out, int out_size, void* d_ws, size_t ws_size,
                              hipStream_t stream) {
  (void)in_sizes; (void)n_in; (void)out_size; (void)ws_size;
  const float* x  = (const float*)d_in[0];
  const float* Wq = (const float*)d_in[1];
  const float* bq = (const float*)d_in[2];
  const float* Wk = (const float*)d_in[3];
  const float* bk = (const float*)d_in[4];
  const float* Wv = (const float*)d_in[5];
  const float* bv = (const float*)d_in[6];
  const float* Wo = (const float*)d_in[7];
  const float* bo = (const float*)d_in[8];
  float* out = (float*)d_out;

  const size_t hnd = (size_t)NH * N_SEQ * DIM;     // 4M elements
  f16* q16  = (f16*)d_ws;
  f16* kT16 = q16 + hnd;
  f16* v16  = kT16 + hnd;
  f16* outs = v16 + hnd;
  f16* x16  = outs + hnd;                          // 512K el
  f16* WT   = x16 + (size_t)N_SEQ * DIM;           // 384K el (3 projs)
  f16* WoT  = WT + (size_t)3 * NH * DIM * DIM;     // 128K el

  prep_kernel<<<4096, 256, 0, stream>>>(x, Wq, Wk, Wv, Wo, x16, WT, WoT);
  qv_proj_kernel<<<512, 256, 0, stream>>>(x16, WT, bq, bv, q16, v16);
  k_proj_kernel<<<256, 256, 0, stream>>>(x16, WT, bk, kT16);
  flash_attn_kernel<<<128, 256, 0, stream>>>(q16, kT16, v16, outs);
  out_proj_kernel<<<256, 256, 0, stream>>>(outs, WoT, bo, out);
}
